// GPT2Shared_17368847745034
// MI455X (gfx1250) — matrix operand order repaired and checked by reading, not compiled, not run on hardware
//
#include <hip/hip_runtime.h>
#include <math.h>

// ---------------------------------------------------------------------------
// Types
// ---------------------------------------------------------------------------
typedef unsigned short u16;
typedef unsigned int   u32;
typedef unsigned long long u64;
typedef __bf16  bf16x16 __attribute__((ext_vector_type(16)));
typedef float   f32x8   __attribute__((ext_vector_type(8)));

__device__ __forceinline__ u16 f2bf(float f) {
    union { float f; u32 u; } c; c.f = f;
    u32 u = c.u;
    if ((u & 0x7fffffffu) > 0x7f800000u) return (u16)((u >> 16) | 0x40); // quiet NaN
    u += 0x7fffu + ((u >> 16) & 1u);                                     // RNE
    return (u16)(u >> 16);
}
__device__ __forceinline__ float bf2f(u16 h) {
    union { u32 u; float f; } c; c.u = ((u32)h) << 16; return c.f;
}

// ---------------------------------------------------------------------------
// f32 -> bf16 elementwise convert
// ---------------------------------------------------------------------------
__global__ void cvt_bf16_kernel(const float* __restrict__ src, u16* __restrict__ dst, int n) {
    int i = blockIdx.x * blockDim.x + threadIdx.x;
    if (i < n) dst[i] = f2bf(src[i]);
}

// ---------------------------------------------------------------------------
// Batched transpose-convert: src [z][K][N] f32 -> dst [z][N][K] bf16
// ---------------------------------------------------------------------------
__global__ void tcvt_bf16_kernel(const float* __restrict__ src, u16* __restrict__ dst,
                                 int K, int N) {
    __shared__ float tile[32][33];
    size_t zoff = (size_t)blockIdx.z * (size_t)K * (size_t)N;
    src += zoff; dst += zoff;
    int kBase = blockIdx.y * 32, nBase = blockIdx.x * 32;
    int tx = threadIdx.x;          // 0..31
    int ty = threadIdx.y;          // 0..7
    #pragma unroll
    for (int i = 0; i < 4; ++i) {
        int k = kBase + ty + i * 8, n = nBase + tx;
        tile[ty + i * 8][tx] = (k < K && n < N) ? src[(size_t)k * N + n] : 0.0f;
    }
    __syncthreads();
    #pragma unroll
    for (int i = 0; i < 4; ++i) {
        int n = nBase + ty + i * 8, k = kBase + tx;
        if (n < N && k < K) dst[(size_t)n * K + k] = f2bf(tile[tx][ty + i * 8]);
    }
}

// ---------------------------------------------------------------------------
// Generic bf16 WMMA GEMM:  C[M,N](bf16) = act(A[M,K] @ Bt[N,K]^T + bias)
//   block tile 128x128, K-chunk 64, DOUBLE-BUFFERED LDS.
//   Global->LDS staging uses CDNA5 async DMA (global_load_async_to_lds_b128,
//   ASYNCcnt-tracked): chunk c+1 streams into the ping-pong buffer while
//   WMMA computes on chunk c; steady state waits asynccnt <= 8 (in-order
//   completion guarantees the previous batch of 8 has landed).
//   EPI: 0 = bias only, 1 = bias+GELU(tanh), 2 = bias+ReLU
//   Requires: M % 128 == 0, K % 64 == 0 (and K >= 64). N guarded (any N).
// ---------------------------------------------------------------------------
template <int EPI>
__global__ __launch_bounds__(256) void gemm_bf16_wmma(
    const u16* __restrict__ A,   // [M][K] row-major
    const u16* __restrict__ Bt,  // [N][K] row-major (B transposed)
    const float* __restrict__ bias, // [N]
    u16* __restrict__ C, int ldc,
    int M, int N, int K) {

    // [2] ping-pong buffers; 64 K + 16 pad keeps 16B alignment per row
    __shared__ __align__(16) u16 As[2][128][80];   // 2 x 20 KB
    __shared__ __align__(16) u16 Bs[2][128][80];   // 2 x 20 KB

    const int tid   = threadIdx.x;
    const int lane  = tid & 31;
    const int wave  = tid >> 5;         // 0..7
    const int waveM = wave & 3;         // 4 waves along M
    const int waveN = wave >> 2;        // 2 waves along N
    const int wm0   = waveM * 32;
    const int wn0   = waveN * 64;
    const int half  = lane >> 4;        // 0/1
    const int l15   = lane & 15;

    const int blockRow = blockIdx.y * 128;
    const int blockCol = blockIdx.x * 128;

    // per-thread staging addresses (4 vectors of 8 bf16 for A, 4 for B)
    u32 aOff[4], bOff[4], ldsA[4], ldsB[4];
    #pragma unroll
    for (int i = 0; i < 4; ++i) {
        int vecId = tid + i * 256;            // 0..1023
        int row   = vecId >> 3;               // 0..127
        int colv  = (vecId & 7) * 8;          // 0..56
        aOff[i] = (u32)(((size_t)(blockRow + row) * (size_t)K + (size_t)colv) * 2u);
        int nb  = blockCol + row; if (nb > N - 1) nb = N - 1;   // clamp (cols >= N never stored)
        bOff[i] = (u32)(((size_t)nb * (size_t)K + (size_t)colv) * 2u);
        ldsA[i] = (u32)(size_t)&As[0][row][colv];               // LDS byte offset (addr[31:0])
        ldsB[i] = (u32)(size_t)&Bs[0][row][colv];
    }
    const u64 aBase = (u64)(size_t)A;
    const u64 bBase = (u64)(size_t)Bt;
    const u32 BUFSZ = 128u * 80u * 2u;        // 20480 bytes per ping-pong slot

    // issue one chunk's 8 async DMA transfers into ping-pong slot (c & 1)
    auto issue_chunk = [&](int c) {
        const u32 kB  = (u32)(c << 7);        // c * 64 elems * 2 bytes
        const u32 sel = (u32)(c & 1) * BUFSZ;
        #pragma unroll
        for (int i = 0; i < 4; ++i) {
            asm volatile("global_load_async_to_lds_b128 %0, %1, %2"
                         :: "v"(ldsA[i] + sel), "v"(aOff[i] + kB), "s"(aBase)
                         : "memory");
            asm volatile("global_load_async_to_lds_b128 %0, %1, %2"
                         :: "v"(ldsB[i] + sel), "v"(bOff[i] + kB), "s"(bBase)
                         : "memory");
        }
    };

    f32x8 acc[2][4];
    #pragma unroll
    for (int i = 0; i < 2; ++i)
        #pragma unroll
        for (int j = 0; j < 4; ++j)
            acc[i][j] = (f32x8)0.0f;

    const int nc = K >> 6;                    // number of 64-wide K chunks
    issue_chunk(0);                           // prologue

    for (int c = 0; c < nc; ++c) {
        if (c + 1 < nc) {
            issue_chunk(c + 1);               // overlap: next chunk DMA in flight
            asm volatile("s_wait_asynccnt 0x8" ::: "memory");  // chunk c landed
        } else {
            asm volatile("s_wait_asynccnt 0x0" ::: "memory");
        }
        __syncthreads();                      // all waves' chunk-c data visible

        const u16 (*Asb)[80] = As[c & 1];
        const u16 (*Bsb)[80] = Bs[c & 1];

        // ---- compute: two WMMA K-steps per LDS chunk ----
        #pragma unroll
        for (int kk = 0; kk < 64; kk += 32) {
            union FA { uint4 u[2]; bf16x16 v; } fa[2];
            union FB { uint4 u[2]; bf16x16 v; } fb[4];
            #pragma unroll
            for (int mt = 0; mt < 2; ++mt) {
                const u16* ap = &Asb[wm0 + mt * 16 + l15][kk + 8 * half];
                fa[mt].u[0] = *(const uint4*)(ap);
                fa[mt].u[1] = *(const uint4*)(ap + 16);
            }
            #pragma unroll
            for (int nt = 0; nt < 4; ++nt) {
                const u16* bp = &Bsb[wn0 + nt * 16 + l15][kk + 8 * half];
                fb[nt].u[0] = *(const uint4*)(bp);
                fb[nt].u[1] = *(const uint4*)(bp + 16);
            }
            #pragma unroll
            for (int mt = 0; mt < 2; ++mt)
                #pragma unroll
                for (int nt = 0; nt < 4; ++nt)
                    acc[mt][nt] = __builtin_amdgcn_wmma_f32_16x16x32_bf16(
                        false, fa[mt].v, false, fb[nt].v,
                        (short)0, acc[mt][nt], false, false);
        }
        __syncthreads();  // all waves done reading buf (c&1) before it is re-filled
    }

    // ---- epilogue: bias + activation, store bf16 ----
    #pragma unroll
    for (int mt = 0; mt < 2; ++mt) {
        #pragma unroll
        for (int nt = 0; nt < 4; ++nt) {
            int gn = blockCol + wn0 + nt * 16 + l15;
            if (gn >= N) continue;
            float b = bias[gn];
            #pragma unroll
            for (int r = 0; r < 8; ++r) {
                int gm = blockRow + wm0 + mt * 16 + r + 8 * half;
                float x = acc[mt][nt][r] + b;
                if (EPI == 1) {
                    float x3 = x * x * x;
                    x = 0.5f * x * (1.0f + tanhf(0.7978845608028654f * (x + 0.044715f * x3)));
                } else if (EPI == 2) {
                    x = fmaxf(x, 0.0f);
                }
                C[(size_t)gm * ldc + gn] = f2bf(x);
            }
        }
    }
}

// ---------------------------------------------------------------------------
// Ensemble layer 3 + accumulate over experts:
//   acc[n][t] += (H2[n][:] @ We3[:,t]) + be3[t]
// ---------------------------------------------------------------------------
__global__ void layer3_acc_kernel(const u16* __restrict__ H2,   // [8192][32] bf16
                                  const float* __restrict__ We3, // [32][5]
                                  const float* __restrict__ be3, // [5]
                                  float* __restrict__ acc) {     // [8192][5]
    int n = blockIdx.x * blockDim.x + threadIdx.x;
    if (n >= 8192) return;
    float h[32];
    const u16* row = H2 + (size_t)n * 32;
    #pragma unroll
    for (int k = 0; k < 32; ++k) h[k] = bf2f(row[k]);
    #pragma unroll
    for (int t = 0; t < 5; ++t) {
        float s = be3[t];
        #pragma unroll
        for (int k = 0; k < 32; ++k) s += h[k] * We3[k * 5 + t];
        acc[(size_t)n * 5 + t] += s;
    }
}

__global__ void zero_f32_kernel(float* __restrict__ p, int n) {
    int i = blockIdx.x * blockDim.x + threadIdx.x;
    if (i < n) p[i] = 0.0f;
}

// mean over T and over E:  out[b][t5] = sum_t acc[b*256+t][t5] / (E*T)
__global__ void reduce_bt_kernel(const float* __restrict__ acc, float* __restrict__ out) {
    __shared__ float sd[256];
    int b = blockIdx.x, t = threadIdx.x;
    for (int j = 0; j < 5; ++j) {
        sd[t] = acc[((size_t)b * 256 + t) * 5 + j];
        __syncthreads();
        for (int s = 128; s > 0; s >>= 1) {
            if (t < s) sd[t] += sd[t + s];
            __syncthreads();
        }
        if (t == 0) out[b * 5 + j] = sd[0] * (1.0f / (32.0f * 256.0f));
        __syncthreads();
    }
}

// ---------------------------------------------------------------------------
// Host-side orchestration
// ---------------------------------------------------------------------------
static void launch_gemm(int epi, const u16* A, const u16* Bt, const float* bias,
                        u16* C, int ldc, int M, int N, int K, hipStream_t s) {
    dim3 grid((N + 127) / 128, M / 128);
    if (epi == 0)      gemm_bf16_wmma<0><<<grid, 256, 0, s>>>(A, Bt, bias, C, ldc, M, N, K);
    else if (epi == 1) gemm_bf16_wmma<1><<<grid, 256, 0, s>>>(A, Bt, bias, C, ldc, M, N, K);
    else               gemm_bf16_wmma<2><<<grid, 256, 0, s>>>(A, Bt, bias, C, ldc, M, N, K);
}

extern "C" void kernel_launch(void* const* d_in, const int* in_sizes, int n_in,
                              void* d_out, int out_size, void* d_ws, size_t ws_size,
                              hipStream_t stream) {
    (void)in_sizes; (void)n_in; (void)out_size; (void)ws_size;

    constexpr int NTOK = 32 * 256;   // 8192 tokens
    constexpr int BASE = 768, C3 = 3 * BASE; // 2304
    constexpr int VID = 768, AUD = 1024, TXT = 768;
    constexpr int E = 32, HID = 32;

    const float* audio = (const float*)d_in[0];
    const float* video = (const float*)d_in[1];
    const float* text  = (const float*)d_in[2];
    const float* Wv = (const float*)d_in[3];  const float* bv = (const float*)d_in[4];
    const float* Wa = (const float*)d_in[5];  const float* ba = (const float*)d_in[6];
    const float* Wt = (const float*)d_in[7];  const float* bt = (const float*)d_in[8];
    const float* Wp = (const float*)d_in[9];  const float* bp = (const float*)d_in[10];
    const float* We1 = (const float*)d_in[11]; const float* be1 = (const float*)d_in[12];
    const float* We2 = (const float*)d_in[13]; const float* be2 = (const float*)d_in[14];
    const float* We3 = (const float*)d_in[15]; const float* be3 = (const float*)d_in[16];
    float* out = (float*)d_out;

    // ---- workspace carve-up ----
    size_t off = 0;
    auto carve = [&](size_t bytes) -> void* {
        void* p = (char*)d_ws + off;
        off += (bytes + 255) & ~(size_t)255;
        return p;
    };
    u16* Xv   = (u16*)carve((size_t)NTOK * VID * 2);
    u16* Xa   = (u16*)carve((size_t)NTOK * AUD * 2);
    u16* Xt   = (u16*)carve((size_t)NTOK * TXT * 2);
    u16* Wv_t = (u16*)carve((size_t)C3 * VID * 2);
    u16* Wa_t = (u16*)carve((size_t)C3 * AUD * 2);
    u16* Wt_t = (u16*)carve((size_t)C3 * TXT * 2);
    u16* Wp_t = (u16*)carve((size_t)BASE * C3 * 2);
    u16* We1t = (u16*)carve((size_t)E * BASE * C3 * 2);
    u16* We2t = (u16*)carve((size_t)E * HID * BASE * 2);
    u16* Y    = (u16*)carve((size_t)NTOK * C3 * 2);
    u16* chunk= (u16*)carve((size_t)NTOK * C3 * 2);
    u16* H1   = (u16*)carve((size_t)NTOK * BASE * 2);
    u16* H2   = (u16*)carve((size_t)NTOK * HID * 2);
    float* acc= (float*)carve((size_t)NTOK * 5 * 4);

    // ---- 1. convert activations to bf16 ----
    {
        int n;
        n = NTOK * VID; cvt_bf16_kernel<<<(n + 255) / 256, 256, 0, stream>>>(video, Xv, n);
        n = NTOK * AUD; cvt_bf16_kernel<<<(n + 255) / 256, 256, 0, stream>>>(audio, Xa, n);
        n = NTOK * TXT; cvt_bf16_kernel<<<(n + 255) / 256, 256, 0, stream>>>(text,  Xt, n);
    }
    // ---- 2. transpose-convert weights to bf16 [N][K] ----
    {
        dim3 blk(32, 8);
        tcvt_bf16_kernel<<<dim3(C3 / 32, VID / 32, 1), blk, 0, stream>>>(Wv, Wv_t, VID, C3);
        tcvt_bf16_kernel<<<dim3(C3 / 32, AUD / 32, 1), blk, 0, stream>>>(Wa, Wa_t, AUD, C3);
        tcvt_bf16_kernel<<<dim3(C3 / 32, TXT / 32, 1), blk, 0, stream>>>(Wt, Wt_t, TXT, C3);
        tcvt_bf16_kernel<<<dim3(BASE / 32, C3 / 32, 1), blk, 0, stream>>>(Wp, Wp_t, C3, BASE);
        tcvt_bf16_kernel<<<dim3(BASE / 32, C3 / 32, E), blk, 0, stream>>>(We1, We1t, C3, BASE);
        tcvt_bf16_kernel<<<dim3(HID / 32, BASE / 32, E), blk, 0, stream>>>(We2, We2t, BASE, HID);
    }
    // ---- 3. adapters + shared projection (chunk = [video | text | audio]) ----
    launch_gemm(1, Xv, Wv_t, bv, Y, C3, NTOK, C3, VID, stream);            // GELU
    launch_gemm(0, Y, Wp_t, bp, chunk + 0, C3, NTOK, BASE, C3, stream);
    launch_gemm(1, Xt, Wt_t, bt, Y, C3, NTOK, C3, TXT, stream);
    launch_gemm(0, Y, Wp_t, bp, chunk + BASE, C3, NTOK, BASE, C3, stream);
    launch_gemm(1, Xa, Wa_t, ba, Y, C3, NTOK, C3, AUD, stream);
    launch_gemm(0, Y, Wp_t, bp, chunk + 2 * BASE, C3, NTOK, BASE, C3, stream);

    // ---- 4. ensemble ----
    zero_f32_kernel<<<(NTOK * 5 + 255) / 256, 256, 0, stream>>>(acc, NTOK * 5);
    for (int e = 0; e < E; ++e) {
        const u16* w1 = We1t + (size_t)e * BASE * C3;
        const u16* w2 = We2t + (size_t)e * HID * BASE;
        launch_gemm(2, chunk, w1, be1 + (size_t)e * BASE, H1, BASE, NTOK, BASE, C3, stream); // ReLU
        launch_gemm(2, H1,    w2, be2 + (size_t)e * HID,  H2, HID,  NTOK, HID,  BASE, stream); // ReLU
        layer3_acc_kernel<<<NTOK / 256, 256, 0, stream>>>(H2, We3 + (size_t)e * HID * 5,
                                                          be3 + (size_t)e * 5, acc);
    }
    // ---- 5. mean over ensemble and time ----
    reduce_bt_kernel<<<32, 256, 0, stream>>>(acc, out);
}
